// AGNNLayer_1262720385540
// MI455X (gfx1250) — matrix operand order repaired and checked by reading, not compiled, not run on hardware
//
#include <hip/hip_runtime.h>
#include <math.h>

#define D 128

typedef __attribute__((ext_vector_type(16))) __bf16 v16bf;
typedef __attribute__((ext_vector_type(8)))  float  v8f;

// ---------------------------------------------------------------------------
// WMMA helpers (CDNA5 bf16 16x16x32, f32 accumulate)
// A layout (16-bit A 16x32, ISA 7.12.2): lane m=lane&15, half=lane>>4 holds
//   K = {half*8 + 0..7} (elems 0..7) and K = {16 + half*8 + 0..7} (elems 8..15)
// B layout (32x16): lane n=lane&15, half holds K = {half*8 + 0..7} and {16 + half*8 + 0..7} (same as A).
// C/D layout: vgpr r, lane l -> M = r + 8*(l>>4), N = l&15.
// ---------------------------------------------------------------------------

static __device__ inline v16bf a_from_f32(const float* __restrict__ row, int kc, int half) {
    const int k0 = kc * 32 + half * 8;
    v16bf a;
#pragma unroll
    for (int i = 0; i < 8; ++i) {
        a[i]     = (__bf16)row[k0 + i];
        a[i + 8] = (__bf16)row[k0 + 16 + i];
    }
    return a;
}

// Same as above, but also spills the f32 values into an LDS tile row so the
// residual add later does not have to re-read e from HBM.
static __device__ inline v16bf a_from_f32_save(const float* __restrict__ row,
                                               float* __restrict__ lds_row,
                                               int kc, int half) {
    const int k0 = kc * 32 + half * 8;
    v16bf a;
#pragma unroll
    for (int i = 0; i < 8; ++i) {
        const float lo = row[k0 + i];
        const float hi = row[k0 + 16 + i];
        lds_row[k0 + i]      = lo;
        lds_row[k0 + 16 + i] = hi;
        a[i]     = (__bf16)lo;
        a[i + 8] = (__bf16)hi;
    }
    return a;
}

static __device__ inline v16bf a_from_bf16(const __bf16* __restrict__ row, int kc, int half) {
    const int k0 = kc * 32 + half * 8;
    v16bf a;
#pragma unroll
    for (int i = 0; i < 8; ++i) {
        a[i]     = row[k0 + i];
        a[i + 8] = row[k0 + 16 + i];
    }
    return a;
}

// B = W.T  (B[k][n] = W[n][k]) -> each lane loads two 8-bf16 runs of a W row (K = half*8 + 0..7, 16 + half*8 + 0..7).
static __device__ inline v16bf b_load(const __bf16* __restrict__ W, int t, int kc, int lane) {
    typedef __attribute__((ext_vector_type(8))) __bf16 v8bf;
    const int n = lane & 15, half = lane >> 4;
    const __bf16* p = W + (size_t)(t * 16 + n) * D + kc * 32 + half * 8;
    const v8bf lo = *(const v8bf*)p;
    const v8bf hi = *(const v8bf*)(p + 16);
    return __builtin_shufflevector(lo, hi, 0, 1, 2, 3, 4, 5, 6, 7, 8, 9, 10, 11, 12, 13, 14, 15);
}

// acc[t] (t = N-tile 0..7) += Xtile(16x128) @ W(128x128).T
// Double-buffered: B fragments for tile t+1 are issued before the 4 WMMAs of
// tile t, so loads stay in flight behind the matrix ops instead of
// serializing load -> s_wait_loadcnt 0 -> wmma.
static __device__ inline void gemm_tile(const v16bf a[4], const __bf16* __restrict__ W,
                                        v8f acc[8], int lane) {
    v16bf b0[4], b1[4];
#pragma unroll
    for (int kc = 0; kc < 4; ++kc) b0[kc] = b_load(W, 0, kc, lane);
#pragma unroll
    for (int t = 0; t < 8; ++t) {
        v16bf(&bc)[4] = (t & 1) ? b1 : b0;
        v16bf(&bn)[4] = (t & 1) ? b0 : b1;
        if (t < 7) {
#pragma unroll
            for (int kc = 0; kc < 4; ++kc) bn[kc] = b_load(W, t + 1, kc, lane);
        }
#pragma unroll
        for (int kc = 0; kc < 4; ++kc) {
            acc[t] = __builtin_amdgcn_wmma_f32_16x16x32_bf16(
                false, a[kc], false, bc[kc], (short)0, acc[t], false, false);
        }
    }
}

static __device__ inline float redsum16(float v) {
#pragma unroll
    for (int m = 1; m < 16; m <<= 1) v += __shfl_xor(v, m, 32);
    return v;
}

// ---------------------------------------------------------------------------
// Kernel: fp32 -> bf16 weight conversion
// ---------------------------------------------------------------------------
__global__ void f32_to_bf16_kernel(const float* __restrict__ s, unsigned short* __restrict__ d, int n) {
    int i = blockIdx.x * blockDim.x + threadIdx.x;
    if (i < n) ((__bf16*)d)[i] = (__bf16)s[i];
}

// ---------------------------------------------------------------------------
// Kernel: time-conditioning MLP (1x128) -> tvec
// ---------------------------------------------------------------------------
__global__ void tmlp_kernel(const float* __restrict__ t_emb,
                            const float* __restrict__ w1, const float* __restrict__ b1,
                            const float* __restrict__ w2, const float* __restrict__ b2,
                            float* __restrict__ out) {
    __shared__ float te[D];
    __shared__ float hid[D];
    const int j = threadIdx.x;
    te[j] = t_emb[j];
    __syncthreads();
    float s = b1[j];
    for (int k = 0; k < D; ++k) s += te[k] * w1[j * D + k];
    hid[j] = fmaxf(s, 0.f);
    __syncthreads();
    float o = b2[j];
    for (int k = 0; k < D; ++k) o += hid[k] * w2[j * D + k];
    out[j] = o;
}

// ---------------------------------------------------------------------------
// Kernel: node projections  Qh/Rh/Vh = h@{Q,R,V}.T ; hagg = h@U.T
// One wave per 16-row tile.
// ---------------------------------------------------------------------------
__global__ __launch_bounds__(256) void node_proj_kernel(
    const float* __restrict__ h,
    const unsigned short* __restrict__ Qw, const unsigned short* __restrict__ Rw,
    const unsigned short* __restrict__ Vw, const unsigned short* __restrict__ Uw,
    float* __restrict__ Qh, float* __restrict__ Rh,
    float* __restrict__ Vh, float* __restrict__ hagg, int N) {
    const int lane = threadIdx.x & 31;
    const int tile = blockIdx.x * (blockDim.x >> 5) + (threadIdx.x >> 5);
    const int base = tile * 16;
    if (base >= N) return;
    const int m = lane & 15, half = lane >> 4, n = lane & 15;

    const float* row = h + (size_t)(base + m) * D;
    v16bf a[4];
#pragma unroll
    for (int kc = 0; kc < 4; ++kc) a[kc] = a_from_f32(row, kc, half);

    auto run = [&](const unsigned short* Wr, float* out) {
        v8f acc[8];
        v8f z = {};
#pragma unroll
        for (int t = 0; t < 8; ++t) acc[t] = z;
        gemm_tile(a, (const __bf16*)Wr, acc, lane);
#pragma unroll
        for (int t = 0; t < 8; ++t)
#pragma unroll
            for (int r = 0; r < 8; ++r)
                out[(size_t)(base + r + half * 8) * D + t * 16 + n] = acc[t][r];
    };
    run(Qw, Qh);
    run(Rw, Rh);
    run(Vw, Vh);
    run(Uw, hagg);  // seeds agg buffer with h@U.T (edge atomics add on top)
}

// ---------------------------------------------------------------------------
// Kernel: fused edge pipeline. One wave per 16-edge tile, 4 waves/block.
// ---------------------------------------------------------------------------
__global__ __launch_bounds__(128) void edge_fused_kernel(
    const float* __restrict__ e, const long long* __restrict__ ei,
    const float* __restrict__ Qh, const float* __restrict__ Rh, const float* __restrict__ Vh,
    const unsigned short* __restrict__ Pw, const unsigned short* __restrict__ W1,
    const unsigned short* __restrict__ W2,
    const float* __restrict__ lng, const float* __restrict__ lnb,
    const float* __restrict__ b1, const float* __restrict__ b2,
    const float* __restrict__ tvec,
    float* __restrict__ e_new, float* __restrict__ hagg, int E_) {
    __shared__ __bf16 stage_s[4][16 * D];  // wave-private re-layout tiles (16 KB)
    __shared__ float  etile_s[4][16 * D];  // wave-private f32 e tiles   (32 KB)

    const int lane = threadIdx.x & 31;
    const int wv   = threadIdx.x >> 5;
    const int base = (blockIdx.x * 4 + wv) * 16;
    if (base >= E_) return;
    const int m = lane & 15, half = lane >> 4, n = lane & 15;
    __bf16* stage = stage_s[wv];
    float*  etile = etile_s[wv];

    // src/dst rows for the C/D layout rows this lane owns (M = r + 8*half)
    int srow[8], drow[8];
#pragma unroll
    for (int r = 0; r < 8; ++r) {
        const int mm = r + half * 8;
        srow[r] = (int)ei[base + mm];
        drow[r] = (int)ei[(size_t)E_ + base + mm];
    }

    // e tile -> bf16 A-chunks (f32 copy parked in LDS for the residual add)
    v16bf a[4];
    {
        const float* row = e + (size_t)(base + m) * D;
        float* lrow = etile + (size_t)m * D;
#pragma unroll
        for (int kc = 0; kc < 4; ++kc) a[kc] = a_from_f32_save(row, lrow, kc, half);
    }

    // e_hat = e@P.T + Qh[src] + Rh[dst]   (gathers served from 192MB L2)
    v8f ehat[8];
    {
        v8f z = {};
#pragma unroll
        for (int t = 0; t < 8; ++t) ehat[t] = z;
    }
    gemm_tile(a, (const __bf16*)Pw, ehat, lane);
#pragma unroll
    for (int t = 0; t < 8; ++t) {
        const int col = t * 16 + n;
#pragma unroll
        for (int r = 0; r < 8; ++r)
            ehat[t][r] += Qh[(size_t)srow[r] * D + col] + Rh[(size_t)drow[r] * D + col];
    }

    // LayerNorm stats: row M lives in the 16 lanes of this half -> shfl reduce
    float mean[8], rstd[8];
#pragma unroll
    for (int r = 0; r < 8; ++r) {
        float s = 0.f, q = 0.f;
#pragma unroll
        for (int t = 0; t < 8; ++t) {
            const float x = ehat[t][r];
            s += x;
            q += x * x;
        }
        s = redsum16(s);
        q = redsum16(q);
        const float mu = s * (1.f / 128.f);
        mean[r] = mu;
        rstd[r] = rsqrtf(q * (1.f / 128.f) - mu * mu + 1e-5f);
    }

    // LN result -> LDS (bf16, row-major) for A-layout reload
#pragma unroll
    for (int t = 0; t < 8; ++t) {
        const int col = t * 16 + n;
        const float g = lng[col], bb = lnb[col];
#pragma unroll
        for (int r = 0; r < 8; ++r)
            stage[(size_t)(r + half * 8) * D + col] =
                (__bf16)((ehat[t][r] - mean[r]) * rstd[r] * g + bb);
    }

    // MLP layer 1: relu(LN @ w1.T + b1)
    v16bf a2[4];
    {
        const __bf16* row = stage + (size_t)m * D;
#pragma unroll
        for (int kc = 0; kc < 4; ++kc) a2[kc] = a_from_bf16(row, kc, half);
    }
    v8f acc[8];
#pragma unroll
    for (int t = 0; t < 8; ++t) {
        const float bv = b1[t * 16 + n];
#pragma unroll
        for (int r = 0; r < 8; ++r) acc[t][r] = bv;
    }
    gemm_tile(a2, (const __bf16*)W1, acc, lane);
#pragma unroll
    for (int t = 0; t < 8; ++t)
#pragma unroll
        for (int r = 0; r < 8; ++r) acc[t][r] = fmaxf(acc[t][r], 0.f);

    // relu -> LDS, reload as A
#pragma unroll
    for (int t = 0; t < 8; ++t) {
        const int col = t * 16 + n;
#pragma unroll
        for (int r = 0; r < 8; ++r)
            stage[(size_t)(r + half * 8) * D + col] = (__bf16)acc[t][r];
    }
    v16bf a3[4];
    {
        const __bf16* row = stage + (size_t)m * D;
#pragma unroll
        for (int kc = 0; kc < 4; ++kc) a3[kc] = a_from_bf16(row, kc, half);
    }

    // MLP layer 2 with (b2 + mlp_t) folded into accumulator, then residual
    v8f acc2[8];
#pragma unroll
    for (int t = 0; t < 8; ++t) {
        const int col = t * 16 + n;
        const float bv = b2[col] + tvec[col];
#pragma unroll
        for (int r = 0; r < 8; ++r) acc2[t][r] = bv;
    }
    gemm_tile(a3, (const __bf16*)W2, acc2, lane);
#pragma unroll
    for (int t = 0; t < 8; ++t)
#pragma unroll
        for (int r = 0; r < 8; ++r) {
            const size_t idx = (size_t)(base + r + half * 8) * D + t * 16 + n;
            e_new[idx] = etile[(size_t)(r + half * 8) * D + t * 16 + n] + acc2[t][r];
        }

    // gated messages: sigmoid(e_hat) * Vh[dst] scatter-added into hagg[src]
#pragma unroll
    for (int t = 0; t < 8; ++t) {
        const int col = t * 16 + n;
#pragma unroll
        for (int r = 0; r < 8; ++r) {
            const float sg  = 1.f / (1.f + __expf(-ehat[t][r]));
            const float msg = sg * Vh[(size_t)drow[r] * D + col];
            atomicAdd(&hagg[(size_t)srow[r] * D + col], msg);
        }
    }
}

// ---------------------------------------------------------------------------
// Kernel: h_new = h + relu(LN(hagg))   one wave per row
// ---------------------------------------------------------------------------
__global__ __launch_bounds__(256) void node_final_kernel(
    const float* __restrict__ h, const float* __restrict__ hagg,
    const float* __restrict__ g, const float* __restrict__ b,
    float* __restrict__ h_new, int N) {
    const int lane = threadIdx.x & 31;
    const int row  = blockIdx.x * (blockDim.x >> 5) + (threadIdx.x >> 5);
    if (row >= N) return;
    const int c = lane * 4;
    const float4 x = *(const float4*)(hagg + (size_t)row * D + c);
    float s = x.x + x.y + x.z + x.w;
    float q = x.x * x.x + x.y * x.y + x.z * x.z + x.w * x.w;
#pragma unroll
    for (int m = 1; m < 32; m <<= 1) {
        s += __shfl_xor(s, m, 32);
        q += __shfl_xor(q, m, 32);
    }
    const float mu   = s * (1.f / 128.f);
    const float rstd = rsqrtf(q * (1.f / 128.f) - mu * mu + 1e-5f);
    const float4 hv = *(const float4*)(h + (size_t)row * D + c);
    float4 o;
    o.x = hv.x + fmaxf((x.x - mu) * rstd * g[c + 0] + b[c + 0], 0.f);
    o.y = hv.y + fmaxf((x.y - mu) * rstd * g[c + 1] + b[c + 1], 0.f);
    o.z = hv.z + fmaxf((x.z - mu) * rstd * g[c + 2] + b[c + 2], 0.f);
    o.w = hv.w + fmaxf((x.w - mu) * rstd * g[c + 3] + b[c + 3], 0.f);
    *(float4*)(h_new + (size_t)row * D + c) = o;
}

// ---------------------------------------------------------------------------
// Host launcher
// ---------------------------------------------------------------------------
extern "C" void kernel_launch(void* const* d_in, const int* in_sizes, int n_in,
                              void* d_out, int out_size, void* d_ws, size_t ws_size,
                              hipStream_t stream) {
    const float*     h       = (const float*)d_in[0];
    const float*     e       = (const float*)d_in[1];
    const long long* ei      = (const long long*)d_in[2];  // int64 (2,E)
    const float*     t_emb   = (const float*)d_in[3];
    const float*     P       = (const float*)d_in[4];
    const float*     Q       = (const float*)d_in[5];
    const float*     R       = (const float*)d_in[6];
    const float*     U       = (const float*)d_in[7];
    const float*     V       = (const float*)d_in[8];
    const float*     ln_e_g  = (const float*)d_in[9];
    const float*     ln_e_b  = (const float*)d_in[10];
    const float*     emlp_w1 = (const float*)d_in[11];
    const float*     emlp_b1 = (const float*)d_in[12];
    const float*     emlp_w2 = (const float*)d_in[13];
    const float*     emlp_b2 = (const float*)d_in[14];
    const float*     tmlp_w1 = (const float*)d_in[15];
    const float*     tmlp_b1 = (const float*)d_in[16];
    const float*     tmlp_w2 = (const float*)d_in[17];
    const float*     tmlp_b2 = (const float*)d_in[18];
    const float*     ln_h_g  = (const float*)d_in[19];
    const float*     ln_h_b  = (const float*)d_in[20];

    const int N = in_sizes[0] / D;
    const int E = in_sizes[1] / D;

    char*  ws  = (char*)d_ws;
    size_t off = 0;
    auto take = [&](size_t bytes) -> void* {
        void* p = ws + off;
        off += (bytes + 255) & ~(size_t)255;
        return p;
    };
    unsigned short* Pw  = (unsigned short*)take((size_t)D * D * 2);
    unsigned short* Qw  = (unsigned short*)take((size_t)D * D * 2);
    unsigned short* Rw  = (unsigned short*)take((size_t)D * D * 2);
    unsigned short* Uw  = (unsigned short*)take((size_t)D * D * 2);
    unsigned short* Vw  = (unsigned short*)take((size_t)D * D * 2);
    unsigned short* W1w = (unsigned short*)take((size_t)D * D * 2);
    unsigned short* W2w = (unsigned short*)take((size_t)D * D * 2);
    float* tvec = (float*)take((size_t)D * 4);
    float* Qh   = (float*)take((size_t)N * D * 4);
    float* Rh   = (float*)take((size_t)N * D * 4);
    float* Vh   = (float*)take((size_t)N * D * 4);
    float* hagg = (float*)take((size_t)N * D * 4);

    // 1) weights -> bf16
    f32_to_bf16_kernel<<<64, 256, 0, stream>>>(P, Pw, D * D);
    f32_to_bf16_kernel<<<64, 256, 0, stream>>>(Q, Qw, D * D);
    f32_to_bf16_kernel<<<64, 256, 0, stream>>>(R, Rw, D * D);
    f32_to_bf16_kernel<<<64, 256, 0, stream>>>(U, Uw, D * D);
    f32_to_bf16_kernel<<<64, 256, 0, stream>>>(V, Vw, D * D);
    f32_to_bf16_kernel<<<64, 256, 0, stream>>>(emlp_w1, W1w, D * D);
    f32_to_bf16_kernel<<<64, 256, 0, stream>>>(emlp_w2, W2w, D * D);

    // 2) time MLP vector
    tmlp_kernel<<<1, D, 0, stream>>>(t_emb, tmlp_w1, tmlp_b1, tmlp_w2, tmlp_b2, tvec);

    // 3) node projections (also seeds hagg = h@U.T)
    const int ntiles = (N + 15) / 16;
    node_proj_kernel<<<(ntiles + 7) / 8, 256, 0, stream>>>(h, Qw, Rw, Vw, Uw, Qh, Rh, Vh, hagg, N);

    // 4) fused edge pipeline
    const int etiles = (E + 15) / 16;
    float* h_new = (float*)d_out;
    float* e_new = (float*)d_out + (size_t)N * D;
    edge_fused_kernel<<<(etiles + 3) / 4, 128, 0, stream>>>(
        e, ei, Qh, Rh, Vh, Pw, W1w, W2w, ln_e_g, ln_e_b, emlp_b1, emlp_b2, tvec,
        e_new, hagg, E);

    // 5) finalize node features
    node_final_kernel<<<(N + 7) / 8, 256, 0, stream>>>(h, hagg, ln_h_g, ln_h_b, h_new, N);
}
